// EncoderLayer_58102317580574
// MI455X (gfx1250) — matrix operand order repaired and checked by reading, not compiled, not run on hardware
//
#include <hip/hip_runtime.h>
#include <hip/hip_bf16.h>

// ---------------------------------------------------------------------------
// Problem constants (fixed by the reference): H=16, M=1024, K=64, V=64,
// FF=4096, B=2, C=2048.  BC = B*C = 4096 token rows.
// ---------------------------------------------------------------------------
#define BC_ROWS   4096
#define MODEL     1024
#define HEADS     16
#define HDIM      64
#define FFDIM     4096
#define SEQ       2048
#define QKV_N     3072          // concat [q|k|v] columns

typedef __attribute__((ext_vector_type(16))) __bf16 v16bf;
typedef __attribute__((ext_vector_type(8)))  float  v8f;

// ---------------------------------------------------------------------------
// WMMA helpers (gfx1250, wave32).  D = A(16x32 bf16) * B(32x16 bf16) + C(f32)
// ---------------------------------------------------------------------------
__device__ __forceinline__ v8f wmma_bf16(v16bf a, v16bf b, v8f c) {
  return __builtin_amdgcn_wmma_f32_16x16x32_bf16(
      /*neg_a=*/false, a, /*neg_b=*/false, b,
      /*c_mod=*/(short)0, c, /*reuse_a=*/false, /*reuse_b=*/false);
}

// A-matrix 16x32 bf16 frag from a row-major tile (row stride = ld elements).
// ISA layout: lane l -> row = l%16, half = l/16;
// VGPR j holds K = (j>>2)*16 + half*8 + (j&3)*2  (+1 in high half).
__device__ __forceinline__ v16bf load_A_frag(const __bf16* t, int ld) {
  int lane = threadIdx.x & 31;
  int row  = lane & 15, half = lane >> 4;
  v16bf a;
#pragma unroll
  for (int j = 0; j < 8; ++j) {
    int kb = ((j >> 2) << 4) + (half << 3) + ((j & 3) << 1);
    a[2 * j]     = t[row * ld + kb];
    a[2 * j + 1] = t[row * ld + kb + 1];
  }
  return a;
}

// B-matrix 32x16 bf16 frag from a tile stored TRANSPOSED (tileT[col][k],
// row stride = ld).  Lane l -> col = l%16, half = l/16;
// VGPR j holds K = (j>>2)*16 + half*8 + (j&3)*2  (+1 in high half).
__device__ __forceinline__ v16bf load_B_frag(const __bf16* tT, int ld) {
  int lane = threadIdx.x & 31;
  int col  = lane & 15, half = lane >> 4;
  v16bf b;
#pragma unroll
  for (int j = 0; j < 8; ++j) {
    int kb = ((j >> 2) << 4) + (half << 3) + ((j & 3) << 1);
    b[2 * j]     = tT[col * ld + kb];
    b[2 * j + 1] = tT[col * ld + kb + 1];
  }
  return b;
}

__device__ __forceinline__ void store_out(float* p, float v)  { *p = v; }
__device__ __forceinline__ void store_out(__bf16* p, float v) { *p = (__bf16)v; }

// ---------------------------------------------------------------------------
// CDNA5 async global->LDS copy (ASYNCcnt path, bypasses VGPRs).
// LDS operand = low 32 bits of the flat shared-aperture address
// ---------------------------------------------------------------------------
__device__ __forceinline__ unsigned lds_off_u32(const void* p) {
  return (unsigned)(unsigned long long)(uintptr_t)p;
}
__device__ __forceinline__ void async_copy_b128(void* lds_dst, const void* gsrc) {
  asm volatile("global_load_async_to_lds_b128 %0, %1, off"
               :
               : "v"(lds_off_u32(lds_dst)),
                 "v"((unsigned long long)(uintptr_t)gsrc)
               : "memory");
}
__device__ __forceinline__ void wait_async_lds() {
  asm volatile("s_wait_asynccnt 0" ::: "memory");
}

// ---------------------------------------------------------------------------
// Elementwise converts
// ---------------------------------------------------------------------------
__global__ void cvt_f32_bf16(const float* __restrict__ s, __bf16* __restrict__ d, int n) {
  int i = blockIdx.x * blockDim.x + threadIdx.x;
  if (i < n) d[i] = (__bf16)s[i];
}

// wq/wk/wv are (H, M, K); build fused bf16 weight W[m][h*64+k | +1024 | +2048]
__global__ void pack_qkv_w(const float* __restrict__ wq, const float* __restrict__ wk,
                           const float* __restrict__ wv, __bf16* __restrict__ d, int n) {
  int i = blockIdx.x * blockDim.x + threadIdx.x;
  if (i >= n) return;
  int k = i % HDIM;
  int m = (i / HDIM) % MODEL;
  int h = i / (HDIM * MODEL);
  size_t base = (size_t)m * QKV_N + h * HDIM + k;
  d[base]             = (__bf16)wq[i];
  d[base + MODEL]     = (__bf16)wk[i];
  d[base + 2 * MODEL] = (__bf16)wv[i];
}

// ---------------------------------------------------------------------------
// Tiled bf16 WMMA GEMM:  C[M x N] = A[M x Kd] * B[Kd x N]  (+bias, relu)
// Block: 256 threads (8 waves), 64x64 tile, BK=32, double-buffered LDS.
// A tile staged via global_load_async_to_lds_b128; B tile staged through
// VGPRs (needs transpose to Bts[col][k]).  Next tile's loads are issued
// before this tile's WMMAs to hide HBM latency.
// Wave w: rows (w>>2)*32 .. +31 (two 16-row WMMA tiles), cols (w&3)*16 .. +15.
// ---------------------------------------------------------------------------
template <typename OutT>
__global__ __launch_bounds__(256) void gemm_bf16_kernel(
    const __bf16* __restrict__ A, const __bf16* __restrict__ Bw,
    OutT* __restrict__ Cc, int Kd, int N,
    const float* __restrict__ bias, int relu) {
  __shared__ __align__(16) __bf16 As[2][64 * 32];
  __shared__ __align__(16) __bf16 Bts[2][64 * 32];   // Bts[col][k]

  int tid   = threadIdx.x;
  size_t bm = (size_t)blockIdx.x * 64;
  size_t bn = (size_t)blockIdx.y * 64;
  int w     = tid >> 5, lane = tid & 31;
  int half  = lane >> 4, nl = lane & 15;
  int m_off = (w >> 2) * 32, n_off = (w & 3) * 16;

  const v8f vz = {0, 0, 0, 0, 0, 0, 0, 0};
  v8f acc0 = vz, acc1 = vz;

  int ar = tid >> 2, ac = (tid & 3) * 8;   // A stage: 64 rows x 32 cols
  int bk = tid >> 3, bc = (tid & 7) * 8;   // B stage: 32 rows x 64 cols

  // ---- prologue: stage tile 0 ----
  async_copy_b128(&As[0][ar * 32 + ac], &A[(bm + ar) * (size_t)Kd + ac]);
  {
    uint4 dv = *(const uint4*)&Bw[(size_t)bk * N + bn + bc];
    __bf16 tmp[8];
    *(uint4*)tmp = dv;
#pragma unroll
    for (int e = 0; e < 8; ++e) Bts[0][(bc + e) * 32 + bk] = tmp[e];
  }

  int cur = 0;
  for (int k0 = 0; k0 < Kd; k0 += 32) {
    wait_async_lds();          // our async writes to As[cur] done
    __syncthreads();           // all waves' stores to Bts[cur] visible too
    int nxt = cur ^ 1;
    bool more = (k0 + 32) < Kd;

    uint4 breg;
    if (more) {
      async_copy_b128(&As[nxt][ar * 32 + ac],
                      &A[(bm + ar) * (size_t)Kd + (k0 + 32) + ac]);
      breg = *(const uint4*)&Bw[(size_t)(k0 + 32 + bk) * N + bn + bc];
    }

    v16bf a0 = load_A_frag(&As[cur][m_off * 32], 32);
    v16bf a1 = load_A_frag(&As[cur][(m_off + 16) * 32], 32);
    v16bf bf = load_B_frag(&Bts[cur][n_off * 32], 32);
    acc0 = wmma_bf16(a0, bf, acc0);
    acc1 = wmma_bf16(a1, bf, acc1);

    if (more) {
      __bf16 tmp[8];
      *(uint4*)tmp = breg;
#pragma unroll
      for (int e = 0; e < 8; ++e) Bts[nxt][(bc + e) * 32 + bk] = tmp[e];
    }
    cur = nxt;
  }

  size_t col = bn + n_off + nl;
  float bb = bias ? bias[col] : 0.0f;
#pragma unroll
  for (int i = 0; i < 8; ++i) {
    size_t r0 = bm + m_off + half * 8 + i;
    float v0 = acc0[i] + bb;
    float v1 = acc1[i] + bb;
    if (relu) { v0 = fmaxf(v0, 0.0f); v1 = fmaxf(v1, 0.0f); }
    store_out(&Cc[r0 * N + col], v0);
    store_out(&Cc[(r0 + 16) * N + col], v1);
  }
}

// ---------------------------------------------------------------------------
// Flash attention.  qkv: (BC_ROWS x 3072) bf16 rows = [q(1024)|k(1024)|v(1024)],
// per head h the 64-wide slice at h*64.  pre out: (BC_ROWS x 1024) bf16 where
// col = h*64 + v  (so out-proj is a plain GEMM against wo flattened (H*V, M)).
// Block: 128 threads (4 waves), 64 queries; wave w owns 16 query rows.
// Q and K tiles staged via async copies; V staged through VGPRs (transposed).
// Key loop: 32 keys/step, S = Q*K^T (4 wmma), online softmax, O += P*V (4 wmma).
// ---------------------------------------------------------------------------
__global__ __launch_bounds__(128) void attn_kernel(
    const __bf16* __restrict__ qkv, __bf16* __restrict__ pre) {
  int qt = blockIdx.x;   // query tile (64 rows)
  int h  = blockIdx.y;
  int b  = blockIdx.z;
  int tid = threadIdx.x;
  int w = tid >> 5, lane = tid & 31;
  int half = lane >> 4, nl = lane & 15;

  __shared__ __align__(16) __bf16 Qs[64 * 64];    // [q][dim]
  __shared__ __align__(16) __bf16 Ks[32 * 64];    // [key][dim]  (== B^T for S)
  __shared__ __align__(16) __bf16 Vt[64 * 32];    // [dim][key]  (== B^T for O)
  __shared__ __align__(16) __bf16 Pb[4 * 16 * 32];// per-wave P scratch

  // stage Q tile (async direct-to-LDS)
#pragma unroll
  for (int it = 0; it < 4; ++it) {
    int idx = it * 128 + tid;
    int r = idx >> 3, ch = (idx & 7) * 8;
    size_t grow = (size_t)(b * SEQ + qt * 64 + r);
    async_copy_b128(&Qs[r * 64 + ch], &qkv[grow * QKV_N + h * HDIM + ch]);
  }
  wait_async_lds();
  __syncthreads();
  v16bf qa0 = load_A_frag(&Qs[(w * 16) * 64], 64);       // dims 0..31
  v16bf qa1 = load_A_frag(&Qs[(w * 16) * 64 + 32], 64);  // dims 32..63

  const v8f vz = {0, 0, 0, 0, 0, 0, 0, 0};
  v8f o[4];
  float m_i[8], l_i[8];
#pragma unroll
  for (int i = 0; i < 8; ++i) { m_i[i] = -1e30f; l_i[i] = 0.0f; }
#pragma unroll
  for (int t = 0; t < 4; ++t) o[t] = vz;

  for (int kt = 0; kt < SEQ / 32; ++kt) {
    __syncthreads();
    // stage K [key][dim] via async; V transposed [dim][key] via VGPRs
#pragma unroll
    for (int it = 0; it < 2; ++it) {
      int idx = it * 128 + tid;
      int r = idx >> 3, ch = (idx & 7) * 8;
      size_t grow = (size_t)(b * SEQ + kt * 32 + r);
      async_copy_b128(&Ks[r * 64 + ch],
                      &qkv[grow * QKV_N + MODEL + h * HDIM + ch]);
      uint4 dv = *(const uint4*)&qkv[grow * QKV_N + 2 * MODEL + h * HDIM + ch];
      __bf16 tmp[8];
      *(uint4*)tmp = dv;
#pragma unroll
      for (int e = 0; e < 8; ++e) Vt[(ch + e) * 32 + r] = tmp[e];
    }
    wait_async_lds();
    __syncthreads();

    // S tiles: keys 0-15 (S0), 16-31 (S1); contraction over 64 dims = 2 wmma.
    v8f S0 = vz, S1 = vz;
    S0 = wmma_bf16(qa0, load_B_frag(&Ks[0], 64), S0);
    S0 = wmma_bf16(qa1, load_B_frag(&Ks[32], 64), S0);
    S1 = wmma_bf16(qa0, load_B_frag(&Ks[16 * 64], 64), S1);
    S1 = wmma_bf16(qa1, load_B_frag(&Ks[16 * 64 + 32], 64), S1);

    // online softmax: rows live in 16-lane halves; shfl_xor 1/2/4/8 reduces.
    float p0[8], p1[8], mx[8], rs[8], alpha[8];
#pragma unroll
    for (int i = 0; i < 8; ++i) {
      p0[i] = S0[i] * 0.125f;   // 1/sqrt(64)
      p1[i] = S1[i] * 0.125f;
      mx[i] = fmaxf(p0[i], p1[i]);
    }
#pragma unroll
    for (int off = 1; off < 16; off <<= 1)
#pragma unroll
      for (int i = 0; i < 8; ++i)
        mx[i] = fmaxf(mx[i], __shfl_xor(mx[i], off, 32));
#pragma unroll
    for (int i = 0; i < 8; ++i) {
      float mn = fmaxf(m_i[i], mx[i]);
      alpha[i] = __expf(m_i[i] - mn);
      p0[i] = __expf(p0[i] - mn);
      p1[i] = __expf(p1[i] - mn);
      rs[i] = p0[i] + p1[i];
      m_i[i] = mn;
    }
#pragma unroll
    for (int off = 1; off < 16; off <<= 1)
#pragma unroll
      for (int i = 0; i < 8; ++i) rs[i] += __shfl_xor(rs[i], off, 32);
#pragma unroll
    for (int i = 0; i < 8; ++i) l_i[i] = l_i[i] * alpha[i] + rs[i];

    // re-shape P from C-layout to A-layout via per-wave LDS scratch
    __bf16* P = &Pb[w * 512];
#pragma unroll
    for (int i = 0; i < 8; ++i) {
      int row = half * 8 + i;
      P[row * 32 + nl]      = (__bf16)p0[i];
      P[row * 32 + 16 + nl] = (__bf16)p1[i];
    }
    __syncthreads();
    v16bf pa = load_A_frag(P, 32);
#pragma unroll
    for (int t = 0; t < 4; ++t) {
#pragma unroll
      for (int i = 0; i < 8; ++i) o[t][i] *= alpha[i];
      v16bf vb = load_B_frag(&Vt[(t * 16) * 32], 32);
      o[t] = wmma_bf16(pa, vb, o[t]);
    }
  }

  float inv[8];
#pragma unroll
  for (int i = 0; i < 8; ++i) inv[i] = 1.0f / l_i[i];
#pragma unroll
  for (int t = 0; t < 4; ++t)
#pragma unroll
    for (int i = 0; i < 8; ++i) {
      int row = qt * 64 + w * 16 + half * 8 + i;
      int col = h * HDIM + t * 16 + nl;
      pre[(size_t)(b * SEQ + row) * MODEL + col] = (__bf16)(o[t][i] * inv[i]);
    }
}

// ---------------------------------------------------------------------------
// Fused residual + LayerNorm over rows of 1024.  out_b may be null.
// ---------------------------------------------------------------------------
__global__ __launch_bounds__(256) void ln_res_kernel(
    const float* __restrict__ A, const float* __restrict__ Bv,
    const float* __restrict__ g, const float* __restrict__ be,
    float* __restrict__ outf, __bf16* __restrict__ outb) {
  int row = blockIdx.x, tid = threadIdx.x;
  __shared__ float red[256];
  const float* pa = A + (size_t)row * MODEL;
  const float* pb = Bv + (size_t)row * MODEL;
  float v[4], s = 0.0f;
#pragma unroll
  for (int j = 0; j < 4; ++j) {
    int idx = tid + j * 256;
    v[j] = pa[idx] + pb[idx];
    s += v[j];
  }
  red[tid] = s;
  __syncthreads();
  for (int st = 128; st > 0; st >>= 1) {
    if (tid < st) red[tid] += red[tid + st];
    __syncthreads();
  }
  float mu = red[0] * (1.0f / MODEL);
  __syncthreads();
  float s2 = 0.0f;
#pragma unroll
  for (int j = 0; j < 4; ++j) { float d = v[j] - mu; s2 += d * d; }
  red[tid] = s2;
  __syncthreads();
  for (int st = 128; st > 0; st >>= 1) {
    if (tid < st) red[tid] += red[tid + st];
    __syncthreads();
  }
  float rstd = rsqrtf(red[0] * (1.0f / MODEL) + 1e-5f);
#pragma unroll
  for (int j = 0; j < 4; ++j) {
    int idx = tid + j * 256;
    float y = (v[j] - mu) * rstd * g[idx] + be[idx];
    outf[(size_t)row * MODEL + idx] = y;
    if (outb) outb[(size_t)row * MODEL + idx] = (__bf16)y;
  }
}

// ---------------------------------------------------------------------------
// Host-side orchestration.  Workspace layout (bytes), with aliasing:
//   s (FFN hidden, 32MB) reuses the [qkv|pre] region; ffn f32 reuses attn f32.
// ---------------------------------------------------------------------------
#define OFF_XB    ((size_t)0)                          // 4096*1024*2  =  8 MB
#define OFF_WQKV  (OFF_XB   + (size_t)BC_ROWS*MODEL*2) // 1024*3072*2  =  6 MB
#define OFF_WO    (OFF_WQKV + (size_t)MODEL*QKV_N*2)   // 1024*1024*2  =  2 MB
#define OFF_W1    (OFF_WO   + (size_t)MODEL*MODEL*2)   // 1024*4096*2  =  8 MB
#define OFF_W2    (OFF_W1   + (size_t)MODEL*FFDIM*2)   // 4096*1024*2  =  8 MB
#define OFF_QKV   (OFF_W2   + (size_t)FFDIM*MODEL*2)   // 4096*3072*2  = 24 MB
#define OFF_PRE   (OFF_QKV  + (size_t)BC_ROWS*QKV_N*2) // 4096*1024*2  =  8 MB
#define OFF_S     OFF_QKV                              // alias (32 MB, qkv+pre dead)
#define OFF_F1    (OFF_PRE  + (size_t)BC_ROWS*MODEL*2) // 4096*1024*4  = 16 MB
#define OFF_H1F   (OFF_F1   + (size_t)BC_ROWS*MODEL*4) // 16 MB
#define OFF_H1B   (OFF_H1F  + (size_t)BC_ROWS*MODEL*4) //  8 MB

extern "C" void kernel_launch(void* const* d_in, const int* in_sizes, int n_in,
                              void* d_out, int out_size, void* d_ws, size_t ws_size,
                              hipStream_t stream) {
  const float* x    = (const float*)d_in[0];
  const float* wq   = (const float*)d_in[1];
  const float* wk   = (const float*)d_in[2];
  const float* wv   = (const float*)d_in[3];
  const float* wo   = (const float*)d_in[4];
  const float* ln1g = (const float*)d_in[5];
  const float* ln1b = (const float*)d_in[6];
  const float* w1   = (const float*)d_in[7];
  const float* b1   = (const float*)d_in[8];
  const float* w2   = (const float*)d_in[9];
  const float* b2   = (const float*)d_in[10];
  const float* ln2g = (const float*)d_in[11];
  const float* ln2b = (const float*)d_in[12];
  float* out = (float*)d_out;
  char*  ws  = (char*)d_ws;

  __bf16* xb    = (__bf16*)(ws + OFF_XB);
  __bf16* wqkvb = (__bf16*)(ws + OFF_WQKV);
  __bf16* wob   = (__bf16*)(ws + OFF_WO);
  __bf16* w1b   = (__bf16*)(ws + OFF_W1);
  __bf16* w2b   = (__bf16*)(ws + OFF_W2);
  __bf16* qkvb  = (__bf16*)(ws + OFF_QKV);
  __bf16* preb  = (__bf16*)(ws + OFF_PRE);
  __bf16* sb    = (__bf16*)(ws + OFF_S);
  float*  f1    = (float*)(ws + OFF_F1);
  float*  h1f   = (float*)(ws + OFF_H1F);
  __bf16* h1b   = (__bf16*)(ws + OFF_H1B);

  int n;
  n = BC_ROWS * MODEL;
  cvt_f32_bf16<<<(n + 255) / 256, 256, 0, stream>>>(x, xb, n);
  n = MODEL * MODEL;      // wo is (H,V,M) == (H*V, M) row-major == 1024x1024
  cvt_f32_bf16<<<(n + 255) / 256, 256, 0, stream>>>(wo, wob, n);
  n = MODEL * FFDIM;
  cvt_f32_bf16<<<(n + 255) / 256, 256, 0, stream>>>(w1, w1b, n);
  n = FFDIM * MODEL;
  cvt_f32_bf16<<<(n + 255) / 256, 256, 0, stream>>>(w2, w2b, n);
  n = HEADS * MODEL * HDIM;
  pack_qkv_w<<<(n + 255) / 256, 256, 0, stream>>>(wq, wk, wv, wqkvb, n);

  // QKV projection: (4096x1024) @ (1024x3072) -> bf16 qkv rows
  gemm_bf16_kernel<__bf16><<<dim3(BC_ROWS / 64, QKV_N / 64), 256, 0, stream>>>(
      xb, wqkvb, qkvb, MODEL, QKV_N, nullptr, 0);

  // Flash attention -> pre (4096 x 1024) bf16, col = h*64+v
  attn_kernel<<<dim3(SEQ / 64, HEADS, 2), 128, 0, stream>>>(qkvb, preb);

  // Output projection: pre @ wo -> attn_out f32
  gemm_bf16_kernel<float><<<dim3(BC_ROWS / 64, MODEL / 64), 256, 0, stream>>>(
      preb, wob, f1, MODEL, MODEL, nullptr, 0);

  // h1 = LN(x + attn_out), emit f32 (residual) + bf16 (FFN GEMM operand)
  ln_res_kernel<<<BC_ROWS, 256, 0, stream>>>(x, f1, ln1g, ln1b, h1f, h1b);

  // FFN1: relu(h1 @ w1 + b1) -> s bf16
  gemm_bf16_kernel<__bf16><<<dim3(BC_ROWS / 64, FFDIM / 64), 256, 0, stream>>>(
      h1b, w1b, sb, MODEL, FFDIM, b1, 1);

  // FFN2: s @ w2 + b2 -> f32
  gemm_bf16_kernel<float><<<dim3(BC_ROWS / 64, MODEL / 64), 256, 0, stream>>>(
      sb, w2b, f1, FFDIM, MODEL, b2, 0);

  // out = LN(h1 + ffn_out)
  ln_res_kernel<<<BC_ROWS, 256, 0, stream>>>(h1f, f1, ln2g, ln2b, out, nullptr);
}
